// RelativePositionMultiHeadAttention_21431886807342
// MI455X (gfx1250) — compile-verified, matrix operand order checked by reading, not run on hardware
//
#include <hip/hip_runtime.h>

// ---------------------------------------------------------------------------
// RelativePositionMultiHeadAttention for MI455X (gfx1250, wave32, WMMA + TDM)
// B=8, L=1024, D=512, H=8, dk=64, buckets=201 (padded to 208)
// ---------------------------------------------------------------------------

typedef _Float16     v16h  __attribute__((ext_vector_type(16)));
typedef float        v8f   __attribute__((ext_vector_type(8)));
typedef unsigned int u32x4 __attribute__((ext_vector_type(4)));
typedef int          i32x8 __attribute__((ext_vector_type(8)));
typedef int          i32x4 __attribute__((ext_vector_type(4)));

#define BATCH 8
#define SEQ   1024
#define DMODEL 512
#define NHEAD 8
#define DK    64
#define NBUCK 201
#define RLD   208   // padded bucket stride

static __device__ __forceinline__ v8f wmma16(v16h a, v16h b, v8f c) {
    // D = A(16x32 f16) * B(32x16 f16) + C(16x16 f32)
    return __builtin_amdgcn_wmma_f32_16x16x32_f16(
        /*neg_a=*/false, a, /*neg_b=*/false, b,
        /*c_mod=*/(short)0, c, /*reuse_a=*/false, /*reuse_b=*/false);
}

// load 16 consecutive f32 and convert to a v16h (addresses are 64B aligned)
static __device__ __forceinline__ v16h load16f_cvt(const float* __restrict__ p) {
    v16h r;
#pragma unroll
    for (int i = 0; i < 4; ++i) {
        float4 t = ((const float4*)p)[i];
        r[4*i+0] = (_Float16)t.x; r[4*i+1] = (_Float16)t.y;
        r[4*i+2] = (_Float16)t.z; r[4*i+3] = (_Float16)t.w;
    }
    return r;
}

// ---------------------------------------------------------------------------
// Tensor Data Mover: async 2D f16 tile load, global -> LDS.
// Descriptor layout per CDNA5 ISA (D# groups 0/1).
//   group0: count=1 | lds_addr[63:32] | global_addr[120:64] | type=2 [127:126]
//   group1: data_size=2B [17:16], tensor_dim0 [79:48], tensor_dim1 [111:80],
//           tile_dim0 [127:112], tile_dim1 [143:128], dim0_stride [207:160]
// remaining groups zero (tensor is 2D). Tracked by TENSORcnt.
// This toolchain exposes the 6-arg builtin (clang-23 / therock-10.0 form).
// ---------------------------------------------------------------------------
static __device__ __forceinline__ void tdm_load_2d_f16(unsigned lds_byte_addr,
                                                       const void* gptr,
                                                       unsigned tensor_d0,
                                                       unsigned tensor_d1,
                                                       unsigned tile_d0,
                                                       unsigned tile_d1,
                                                       unsigned d0_stride) {
    const unsigned long long ga = (unsigned long long)(uintptr_t)gptr;
    u32x4 g0;
    g0[0] = 1u;                                        // count=1, no gather
    g0[1] = lds_byte_addr;                             // lds_addr
    g0[2] = (unsigned)ga;                              // global_addr[31:0]
    g0[3] = (unsigned)(ga >> 32) | 0x80000000u;        // addr[56:32] | type=2
    i32x8 g1;
    g1[0] = 0x10000;                                   // data_size = 1 (2 bytes)
    g1[1] = (int)((tensor_d0 & 0xFFFFu) << 16);        // tensor_dim0[15:0]
    g1[2] = (int)((tensor_d0 >> 16) | ((tensor_d1 & 0xFFFFu) << 16));
    g1[3] = (int)((tensor_d1 >> 16) | (tile_d0 << 16));// tile_dim0
    g1[4] = (int)tile_d1;                              // tile_dim1 (tile_dim2=0)
    g1[5] = (int)d0_stride;                            // tensor_dim0_stride[31:0]
    g1[6] = 0;
    g1[7] = 0;
    const i32x4 z4 = {0, 0, 0, 0};
    const i32x8 z8 = {0, 0, 0, 0, 0, 0, 0, 0};
    __builtin_amdgcn_tensor_load_to_lds(g0, g1, z4, z4, z8, 0);
}

// ---------------------------------------------------------------------------
// Projection GEMM: Y[M,N] = X[M,K](f32) * W[N,K](f32)^T, f16 out.
// TRANS==1 stores Y transposed per-batch as [B, N, L] (used for V).
// Block 256 = 8 waves: 4 M-tiles x 2 N-tiles -> 64x128 block tile.
// ---------------------------------------------------------------------------
template <int TRANS>
__global__ __launch_bounds__(256) void proj_gemm(const float* __restrict__ X,
                                                 const float* __restrict__ W,
                                                 _Float16*    __restrict__ Y) {
    const int K = DMODEL, N = DMODEL;
    const int w    = threadIdx.x >> 5;
    const int lane = threadIdx.x & 31;
    const int ln   = lane & 15;
    const int hi   = lane >> 4;           // 0: K 0..15, 1: K 16..31 of chunk
    const int mbase = blockIdx.x * 64  + (w & 3) * 16;
    const int nbase = blockIdx.y * 128 + (w >> 2) * 64;

    v8f acc[4] = {};
    for (int kc = 0; kc < K; kc += 32) {
        v16h av = load16f_cvt(X + (size_t)(mbase + ln) * K + kc + hi * 16);
#pragma unroll
        for (int c = 0; c < 4; ++c) {
            v16h bv = load16f_cvt(W + (size_t)(nbase + c*16 + ln) * K + kc + hi * 16);
            acc[c] = wmma16(av, bv, acc[c]);
        }
    }
#pragma unroll
    for (int c = 0; c < 4; ++c) {
#pragma unroll
        for (int r = 0; r < 8; ++r) {
            const int row = mbase + r + hi * 8;      // C layout: lanes 16-31 hold M+8
            const int col = nbase + c * 16 + ln;
            if (TRANS) {                             // V: store as [B, D, L]
                const int b = row >> 10, tok = row & 1023;
                Y[((size_t)(b * DMODEL + col)) * SEQ + tok] = (_Float16)acc[c][r];
            } else {
                Y[(size_t)row * N + col] = (_Float16)acc[c][r];
            }
        }
    }
}

// ---------------------------------------------------------------------------
// Output GEMM: Y[M,N](f32) = A[M,K](f16) * W[N,K](f32)^T  (residual/LN later)
// ---------------------------------------------------------------------------
__global__ __launch_bounds__(256) void out_gemm(const _Float16* __restrict__ A,
                                                const float*    __restrict__ W,
                                                float*          __restrict__ Y) {
    const int K = DMODEL, N = DMODEL;
    const int w    = threadIdx.x >> 5;
    const int lane = threadIdx.x & 31;
    const int ln   = lane & 15;
    const int hi   = lane >> 4;
    const int mbase = blockIdx.x * 64  + (w & 3) * 16;
    const int nbase = blockIdx.y * 128 + (w >> 2) * 64;

    v8f acc[4] = {};
    for (int kc = 0; kc < K; kc += 32) {
        v16h av = *(const v16h*)(A + (size_t)(mbase + ln) * K + kc + hi * 16);
#pragma unroll
        for (int c = 0; c < 4; ++c) {
            v16h bv = load16f_cvt(W + (size_t)(nbase + c*16 + ln) * K + kc + hi * 16);
            acc[c] = wmma16(av, bv, acc[c]);
        }
    }
#pragma unroll
    for (int c = 0; c < 4; ++c)
#pragma unroll
        for (int r = 0; r < 8; ++r)
            Y[(size_t)(mbase + r + hi * 8) * N + nbase + c * 16 + ln] = acc[c][r];
}

// ---------------------------------------------------------------------------
// Position table GEMM: R[b,h,q,t] = Qh[b,q,h*64:+64] . rel_emb[t,:]
// One wave = 16 q rows x 13 bucket chunks (201 padded to 208). K = 64.
// ---------------------------------------------------------------------------
__global__ __launch_bounds__(256) void rel_gemm(const _Float16* __restrict__ Qh,
                                                const float*    __restrict__ rel,
                                                _Float16*       __restrict__ R) {
    const int w    = threadIdx.x >> 5;
    const int lane = threadIdx.x & 31;
    const int ln   = lane & 15;
    const int hi   = lane >> 4;
    const int bh = blockIdx.y;                 // b*8 + h
    const int b  = bh >> 3, h = bh & 7;
    const int q0 = blockIdx.x * 128 + w * 16;

    const _Float16* abase = Qh + ((size_t)(b * SEQ + q0 + ln)) * DMODEL + h * DK + hi * 16;
    v16h a0 = *(const v16h*)(abase);
    v16h a1 = *(const v16h*)(abase + 32);

#pragma unroll 1
    for (int c = 0; c < 13; ++c) {
        int t = c * 16 + ln; if (t > NBUCK - 1) t = NBUCK - 1;   // clamp pad cols
        v16h b0 = load16f_cvt(rel + (size_t)t * DK + 0  + hi * 16);
        v16h b1 = load16f_cvt(rel + (size_t)t * DK + 32 + hi * 16);
        v8f acc = {};
        acc = wmma16(a0, b0, acc);
        acc = wmma16(a1, b1, acc);
#pragma unroll
        for (int r = 0; r < 8; ++r) {
            const int q = q0 + r + hi * 8;
            R[((size_t)bh * SEQ + q) * RLD + c * 16 + ln] = (_Float16)acc[r];
        }
    }
}

// ---------------------------------------------------------------------------
// Flash attention with relative-position gather, TDM-staged K/V tiles.
// Block 128 = 4 waves; wave owns 16 q rows; 32-token K/V chunks.
// Wave 0 issues tensor_load_to_lds for tile i+1 (double buffered) while all
// waves compute on tile i out of LDS; sync = s_wait_tensorcnt + barrier.
// ---------------------------------------------------------------------------
__global__ __launch_bounds__(128) void flash_attn(const _Float16* __restrict__ Qh,
                                                  const _Float16* __restrict__ Kh,
                                                  const _Float16* __restrict__ Vt,
                                                  const _Float16* __restrict__ Rpos,
                                                  _Float16*       __restrict__ O16) {
    const int w    = threadIdx.x >> 5;
    const int lane = threadIdx.x & 31;
    const int ln   = lane & 15;
    const int hi   = lane >> 4;
    const int b = blockIdx.z, h = blockIdx.y;
    const int bh = b * NHEAD + h;
    const int q0 = blockIdx.x * 64 + w * 16;

    // Double-buffered DMA tiles + P bounce buffer (20 KB of the 320 KB WGP LDS)
    __shared__ __align__(128) _Float16 ktile[2][32][64];   // [token][dk]
    __shared__ __align__(128) _Float16 vtile[2][64][32];   // [dk][token]
    __shared__ __align__(64)  _Float16 pbuf[4][16][32];

    const _Float16* kglob = Kh + ((size_t)(b * SEQ)) * DMODEL + h * DK;
    const _Float16* vglob = Vt + ((size_t)(b * DMODEL + h * DK)) * SEQ;

    // resident Q tile: 16 rows x 64 dk (A layout, two K=32 chunks)
    const _Float16* qbase = Qh + ((size_t)(b * SEQ + q0 + ln)) * DMODEL + h * DK + hi * 16;
    const v16h qa0 = *(const v16h*)(qbase);
    const v16h qa1 = *(const v16h*)(qbase + 32);

    v8f o0 = {}, o1 = {}, o2 = {}, o3 = {};
    float mrow[8], lrow[8];
#pragma unroll
    for (int r = 0; r < 8; ++r) { mrow[r] = -1e30f; lrow[r] = 0.f; }

    const _Float16* rrow_base = Rpos + ((size_t)bh * SEQ) * RLD;

    // prologue: DMA tile 0
    if (w == 0) {
        tdm_load_2d_f16((unsigned)(uintptr_t)&ktile[0][0][0], kglob,
                        DK, 32, DK, 32, DMODEL);
        tdm_load_2d_f16((unsigned)(uintptr_t)&vtile[0][0][0], vglob,
                        32, DK, 32, DK, SEQ);
    }

    for (int kb = 0; kb < SEQ; kb += 32) {
        const int buf = (kb >> 5) & 1;
        if (w == 0) __builtin_amdgcn_s_wait_tensorcnt(0);   // tile `buf` landed
        __syncthreads();
        if (w == 0 && kb + 32 < SEQ) {                      // prefetch tile i+1
            tdm_load_2d_f16((unsigned)(uintptr_t)&ktile[buf ^ 1][0][0],
                            kglob + (size_t)(kb + 32) * DMODEL, DK, 32, DK, 32, DMODEL);
            tdm_load_2d_f16((unsigned)(uintptr_t)&vtile[buf ^ 1][0][0],
                            vglob + kb + 32, 32, DK, 32, DK, SEQ);
        }
        const _Float16 (*kt)[64] = ktile[buf];
        const _Float16 (*vt)[32] = vtile[buf];

        // ---- S = Q * K^T for 32 key columns (two 16-col tiles) ----
        v8f s0 = {}, s1 = {};
        {
            v16h b00 = *(const v16h*)(&kt[ln][hi * 16]);
            v16h b01 = *(const v16h*)(&kt[ln][32 + hi * 16]);
            s0 = wmma16(qa0, b00, s0);
            s0 = wmma16(qa1, b01, s0);
            v16h b10 = *(const v16h*)(&kt[16 + ln][hi * 16]);
            v16h b11 = *(const v16h*)(&kt[16 + ln][32 + hi * 16]);
            s1 = wmma16(qa0, b10, s1);
            s1 = wmma16(qa1, b11, s1);
        }

        // ---- position gather + scale + online softmax ----
        float p0[8], p1[8];
#pragma unroll
        for (int r = 0; r < 8; ++r) {
            const int q  = q0 + r + hi * 8;
            const int k0 = kb + ln, k1 = kb + 16 + ln;
            int d0 = q - k0; d0 = d0 < -100 ? -100 : (d0 > 100 ? 100 : d0);
            int d1 = q - k1; d1 = d1 < -100 ? -100 : (d1 > 100 ? 100 : d1);
            const _Float16* rr = rrow_base + (size_t)q * RLD;
            float a0 = (s0[r] + (float)rr[d0 + 100]) * 0.125f;   // 1/sqrt(64)
            float a1 = (s1[r] + (float)rr[d1 + 100]) * 0.125f;

            float tmax = fmaxf(a0, a1);
#pragma unroll
            for (int s = 1; s < 16; s <<= 1) tmax = fmaxf(tmax, __shfl_xor(tmax, s, 32));
            const float nm   = fmaxf(mrow[r], tmax);
            const float corr = __expf(mrow[r] - nm);
            mrow[r] = nm;
            const float e0 = __expf(a0 - nm), e1 = __expf(a1 - nm);
            float rs = e0 + e1;
#pragma unroll
            for (int s = 1; s < 16; s <<= 1) rs += __shfl_xor(rs, s, 32);
            lrow[r] = lrow[r] * corr + rs;
            o0[r] *= corr; o1[r] *= corr; o2[r] *= corr; o3[r] *= corr;
            p0[r] = e0; p1[r] = e1;
        }

        // ---- P (C layout) -> LDS row-major -> reload in A layout ----
#pragma unroll
        for (int r = 0; r < 8; ++r) {
            const int m = r + hi * 8;
            pbuf[w][m][ln]      = (_Float16)p0[r];
            pbuf[w][m][16 + ln] = (_Float16)p1[r];
        }
        const v16h pa = *(const v16h*)(&pbuf[w][ln][hi * 16]);

        // ---- O += P * V  (vtile is [dk][token]: columns K-contiguous) ----
        o0 = wmma16(pa, *(const v16h*)(&vt[0  + ln][hi * 16]), o0);
        o1 = wmma16(pa, *(const v16h*)(&vt[16 + ln][hi * 16]), o1);
        o2 = wmma16(pa, *(const v16h*)(&vt[32 + ln][hi * 16]), o2);
        o3 = wmma16(pa, *(const v16h*)(&vt[48 + ln][hi * 16]), o3);

        __syncthreads();   // everyone done with tile `buf` before it is reused
    }

    // ---- normalize and store f16 tile ----
#pragma unroll
    for (int r = 0; r < 8; ++r) {
        const int q = q0 + r + hi * 8;
        const float inv = 1.0f / lrow[r];
        _Float16* orow = O16 + ((size_t)(b * SEQ + q)) * DMODEL + h * DK;
        orow[0  + ln] = (_Float16)(o0[r] * inv);
        orow[16 + ln] = (_Float16)(o1[r] * inv);
        orow[32 + ln] = (_Float16)(o2[r] * inv);
        orow[48 + ln] = (_Float16)(o3[r] * inv);
    }
}

// ---------------------------------------------------------------------------
// Residual add + LayerNorm over D=512. One block (128 thr) per row.
// ---------------------------------------------------------------------------
__global__ __launch_bounds__(128) void ln_kernel(const float* __restrict__ Y,
                                                 const float* __restrict__ resid,
                                                 const float* __restrict__ gamma,
                                                 const float* __restrict__ beta,
                                                 float*       __restrict__ out) {
    const int row = blockIdx.x;
    const int tid = threadIdx.x;
    const float* y = Y     + (size_t)row * DMODEL;
    const float* r = resid + (size_t)row * DMODEL;

    float x[4], s = 0.f, ss = 0.f;
#pragma unroll
    for (int i = 0; i < 4; ++i) {
        const int c = tid + i * 128;
        x[i] = y[c] + r[c];
        s += x[i]; ss += x[i] * x[i];
    }
#pragma unroll
    for (int m = 1; m < 32; m <<= 1) { s += __shfl_xor(s, m, 32); ss += __shfl_xor(ss, m, 32); }

    __shared__ float sb[4], ssb[4];
    const int w = tid >> 5;
    if ((tid & 31) == 0) { sb[w] = s; ssb[w] = ss; }
    __syncthreads();
    s  = sb[0]  + sb[1]  + sb[2]  + sb[3];
    ss = ssb[0] + ssb[1] + ssb[2] + ssb[3];

    const float mu  = s * (1.0f / DMODEL);
    const float var = ss * (1.0f / DMODEL) - mu * mu;
    const float rs  = rsqrtf(var + 1e-6f);
#pragma unroll
    for (int i = 0; i < 4; ++i) {
        const int c = tid + i * 128;
        out[(size_t)row * DMODEL + c] = (x[i] - mu) * rs * gamma[c] + beta[c];
    }
}

// ---------------------------------------------------------------------------
// Host orchestration. Workspace layout (bytes):
//   Qh16 @ 0         (8 MiB)   [B,L,D] f16
//   Kh16 @ 8388608   (8 MiB)   [B,L,D] f16
//   Vt16 @ 16777216  (8 MiB)   [B,D,L] f16  (transposed)
//   O16  @ 25165824  (8 MiB)   [B,L,D] f16
//   R    @ 33554432  (26 MiB)  [B*H,L,208] f16
//   Y32  @ 60817408  (16 MiB)  [B,L,D] f32  pre-LN
// ---------------------------------------------------------------------------
extern "C" void kernel_launch(void* const* d_in, const int* in_sizes, int n_in,
                              void* d_out, int out_size, void* d_ws, size_t ws_size,
                              hipStream_t stream) {
    const float* q     = (const float*)d_in[0];
    const float* k     = (const float*)d_in[1];
    const float* v     = (const float*)d_in[2];
    const float* Wq    = (const float*)d_in[3];
    const float* Wk    = (const float*)d_in[4];
    const float* Wv    = (const float*)d_in[5];
    const float* Wfc   = (const float*)d_in[6];
    const float* rel   = (const float*)d_in[7];
    const float* gamma = (const float*)d_in[8];
    const float* beta  = (const float*)d_in[9];

    char* ws = (char*)d_ws;
    _Float16* Qh  = (_Float16*)(ws + 0);
    _Float16* Kh  = (_Float16*)(ws + 8388608);
    _Float16* Vt  = (_Float16*)(ws + 16777216);
    _Float16* O16 = (_Float16*)(ws + 25165824);
    _Float16* R   = (_Float16*)(ws + 33554432);
    float*    Y32 = (float*)   (ws + 60817408);

    const dim3 gg(128, 4), gb(256);
    proj_gemm<0><<<gg, gb, 0, stream>>>(q, Wq, Qh);
    proj_gemm<0><<<gg, gb, 0, stream>>>(k, Wk, Kh);
    proj_gemm<1><<<gg, gb, 0, stream>>>(v, Wv, Vt);

    rel_gemm<<<dim3(8, 64), 256, 0, stream>>>(Qh, rel, R);

    flash_attn<<<dim3(16, NHEAD, BATCH), 128, 0, stream>>>(Qh, Kh, Vt, R, O16);

    out_gemm<<<gg, gb, 0, stream>>>(O16, Wfc, Y32);

    ln_kernel<<<BATCH * SEQ, 128, 0, stream>>>(Y32, q, gamma, beta, (float*)d_out);
}
